// MultiHeadAttention_23983097381592
// MI455X (gfx1250) — matrix operand order repaired and checked by reading, not compiled, not run on hardware
//
#include <hip/hip_runtime.h>

typedef __bf16 bf16;
typedef __attribute__((ext_vector_type(16))) __bf16 v16bf;
typedef __attribute__((ext_vector_type(8)))  float  v8f;

#define S_LEN 2048
#define DMODEL 1024
#define NHEADS 16
#define DK 64

__device__ __forceinline__ v8f wmma_bf16(v16bf a, v16bf b, v8f c) {
  return __builtin_amdgcn_wmma_f32_16x16x32_bf16(false, a, false, b, (short)0, c, false, false);
}

// A-frag (16x32 bf16, row-major base [M][K], leading dim ld elements)
// lane: m = l&15, koff = (l>=16)*8 ; elems 0..7 = K[koff..koff+8), elems 8..15 = K[16+koff..24+koff)
__device__ __forceinline__ v16bf load_afrag(const bf16* base, int ld) {
  int lane = threadIdx.x & 31;
  int m = lane & 15;
  int koff = (lane >> 4) << 3;
  const bf16* p = base + m * ld + koff;
  v16bf f;
  ((float4*)&f)[0] = *(const float4*)(p);
  ((float4*)&f)[1] = *(const float4*)(p + 16);
  return f;
}

// B-frag (32x16 bf16) from COLUMN-major storage: base points at [n=0][k=0] of Bt[n][k],
// ld = elements between columns. lane: n = l&15, koff = (l>=16)*8 ; elems 0..7 = K[koff..koff+8), elems 8..15 = K[16+koff..24+koff)
__device__ __forceinline__ v16bf load_bfrag(const bf16* base, int ld) {
  int lane = threadIdx.x & 31;
  int n = lane & 15;
  int koff = (lane >> 4) << 3;
  const bf16* p = base + n * ld + koff;
  v16bf f;
  ((float4*)&f)[0] = *(const float4*)(p);
  ((float4*)&f)[1] = *(const float4*)(p + 16);
  return f;
}

__device__ __forceinline__ float rowmax16(float v) {
  v = fmaxf(v, __shfl_xor(v, 1));
  v = fmaxf(v, __shfl_xor(v, 2));
  v = fmaxf(v, __shfl_xor(v, 4));
  v = fmaxf(v, __shfl_xor(v, 8));
  return v;
}
__device__ __forceinline__ float rowsum16(float v) {
  v += __shfl_xor(v, 1);
  v += __shfl_xor(v, 2);
  v += __shfl_xor(v, 4);
  v += __shfl_xor(v, 8);
  return v;
}

// ---------------- conversion kernels ----------------
__global__ void conv_f32_bf16(const float* __restrict__ in, bf16* __restrict__ out, int n) {
  int i = blockIdx.x * 256 + threadIdx.x;
  if (i < n) out[i] = (bf16)in[i];
}

// Wt[n][k] = W[k][n], bf16 (1024x1024)
__global__ void transpose_conv(const float* __restrict__ in, bf16* __restrict__ out) {
  int i = blockIdx.x * 256 + threadIdx.x;
  int k = i & (DMODEL - 1);
  int n = i >> 10;
  out[(size_t)n * DMODEL + k] = (bf16)in[(size_t)k * DMODEL + n];
}

// ---------------- GEMM: C[M=8192][N=1024] = A[M][K=1024] @ Bt^T + bias ----------------
// MODE 0: write bf16 [b*16+h][s][dk]   (Q, K)
// MODE 1: write bf16 [b*16+h][dk][s]   (V transposed)
// MODE 2: write fp32 row-major [M][N]  (final output)
template <int MODE>
__global__ void gemm_kernel(const bf16* __restrict__ A, const bf16* __restrict__ Bt,
                            const float* __restrict__ bias, void* __restrict__ outv) {
  __shared__ __align__(16) bf16 sA[128 * 32];
  __shared__ __align__(16) bf16 sB[128 * 32];
  const int K = DMODEL;
  int tid = threadIdx.x;
  int wave = tid >> 5;
  int wm = wave & 3;       // 4 m-waves * 32 rows
  int wn = wave >> 2;      // 2 n-waves * 64 cols
  int bm = blockIdx.x, bn = blockIdx.y;

  int lrow = tid >> 1;           // 0..127
  int lseg = (tid & 1) << 4;     // 0 or 16 elements
  const bf16* Ag = A + (size_t)(bm * 128 + lrow) * K + lseg;
  const bf16* Bg = Bt + (size_t)(bn * 128 + lrow) * K + lseg;
  bf16* sAp = sA + lrow * 32 + lseg;
  bf16* sBp = sB + lrow * 32 + lseg;

  v8f acc[2][4];
#pragma unroll
  for (int mi = 0; mi < 2; mi++)
#pragma unroll
    for (int ni = 0; ni < 4; ni++)
#pragma unroll
      for (int r = 0; r < 8; r++) acc[mi][ni][r] = 0.0f;

  for (int kt = 0; kt < K; kt += 32) {
    __syncthreads();
    *(float4*)sAp       = *(const float4*)(Ag + kt);
    *(float4*)(sAp + 8) = *(const float4*)(Ag + kt + 8);
    *(float4*)sBp       = *(const float4*)(Bg + kt);
    *(float4*)(sBp + 8) = *(const float4*)(Bg + kt + 8);
    __syncthreads();

    v16bf a0 = load_afrag(sA + (wm * 32 + 0) * 32, 32);
    v16bf a1 = load_afrag(sA + (wm * 32 + 16) * 32, 32);
#pragma unroll
    for (int ni = 0; ni < 4; ni++) {
      v16bf bfr = load_bfrag(sB + (wn * 64 + ni * 16) * 32, 32);
      acc[0][ni] = wmma_bf16(a0, bfr, acc[0][ni]);
      acc[1][ni] = wmma_bf16(a1, bfr, acc[1][ni]);
    }
  }

  int lane = tid & 31, ln = lane & 15, half = lane >> 4;
#pragma unroll
  for (int ni = 0; ni < 4; ni++) {
    int gn = bn * 128 + wn * 64 + ni * 16 + ln;
    float bv = bias[gn];
#pragma unroll
    for (int mi = 0; mi < 2; mi++) {
#pragma unroll
      for (int r = 0; r < 8; r++) {
        int gm = bm * 128 + wm * 32 + mi * 16 + half * 8 + r;
        float val = acc[mi][ni][r] + bv;
        if (MODE == 2) {
          ((float*)outv)[(size_t)gm * DMODEL + gn] = val;
        } else {
          int b = gm >> 11, s = gm & (S_LEN - 1);
          int h = gn >> 6, d = gn & (DK - 1);
          bf16* o = (bf16*)outv;
          if (MODE == 0)
            o[(((size_t)(b * NHEADS + h)) * S_LEN + s) * DK + d] = (bf16)val;
          else
            o[(((size_t)(b * NHEADS + h)) * DK + d) * S_LEN + s] = (bf16)val;
        }
      }
    }
  }
}

// ---------------- flash attention: 1 wave = 16 query rows ----------------
// Q,K: [bh][s][dk] bf16 ; Vt: [bh][dk][s] bf16 ; AO: [b][s][H*dk] bf16
__global__ void attn_kernel(const bf16* __restrict__ Q, const bf16* __restrict__ Kb,
                            const bf16* __restrict__ Vt, const int* __restrict__ mask,
                            bf16* __restrict__ AO) {
  __shared__ __align__(16) bf16 lds_p[8][16 * 32];
  int wave = threadIdx.x >> 5;
  int lane = threadIdx.x & 31;
  int ln = lane & 15, half = lane >> 4;
  int bh = blockIdx.y;
  int b = bh >> 4, h = bh & 15;
  int q0 = (blockIdx.x * 8 + wave) * 16;
  const bf16* Qbh = Q + (size_t)bh * S_LEN * DK;
  const bf16* Kbh = Kb + (size_t)bh * S_LEN * DK;
  const bf16* Vbh = Vt + (size_t)bh * DK * S_LEN;
  const int* mb = mask + b * S_LEN;

  v16bf qa0 = load_afrag(Qbh + q0 * DK, DK);
  v16bf qa1 = load_afrag(Qbh + q0 * DK + 32, DK);

  float mrun[8], lrun[8];
  v8f o[4];
#pragma unroll
  for (int r = 0; r < 8; r++) { mrun[r] = -3.0e38f; lrun[r] = 0.0f; }
#pragma unroll
  for (int t = 0; t < 4; t++)
#pragma unroll
    for (int r = 0; r < 8; r++) o[t][r] = 0.0f;

  bf16* lp = lds_p[wave];

  for (int kt = 0; kt < S_LEN; kt += 32) {
    v8f z;
#pragma unroll
    for (int r = 0; r < 8; r++) z[r] = 0.0f;
    // scores = Q @ K^T : two 16x16 tiles (keys kt..+16, kt+16..+32), K reduction = 64 = 2 wmma
    v16bf kb00 = load_bfrag(Kbh + kt * DK, DK);
    v16bf kb01 = load_bfrag(Kbh + kt * DK + 32, DK);
    v16bf kb10 = load_bfrag(Kbh + (kt + 16) * DK, DK);
    v16bf kb11 = load_bfrag(Kbh + (kt + 16) * DK + 32, DK);
    v8f s0 = wmma_bf16(qa1, kb01, wmma_bf16(qa0, kb00, z));
    v8f s1 = wmma_bf16(qa1, kb11, wmma_bf16(qa0, kb10, z));

    int mv0 = mb[kt + ln];
    int mv1 = mb[kt + 16 + ln];
    float p0[8], p1[8];
#pragma unroll
    for (int r = 0; r < 8; r++) {
      float a = mv0 ? s0[r] * 0.125f : -1.0e9f;
      float c = mv1 ? s1[r] * 0.125f : -1.0e9f;
      float rm = rowmax16(fmaxf(a, c));
      float mn = fmaxf(mrun[r], rm);
      p0[r] = __expf(a - mn);
      p1[r] = __expf(c - mn);
      float rs = rowsum16(p0[r] + p1[r]);
      float corr = __expf(mrun[r] - mn);
      lrun[r] = lrun[r] * corr + rs;
      mrun[r] = mn;
      o[0][r] *= corr; o[1][r] *= corr; o[2][r] *= corr; o[3][r] *= corr;
    }
    // transpose P (C-layout -> A-frag layout) through wave-private LDS
#pragma unroll
    for (int r = 0; r < 8; r++) {
      lp[(half * 8 + r) * 32 + ln]      = (bf16)p0[r];
      lp[(half * 8 + r) * 32 + 16 + ln] = (bf16)p1[r];
    }
    asm volatile("s_wait_dscnt 0" ::: "memory");  // wave-local LDS RAW
    v16bf pa = load_afrag(lp, 32);
#pragma unroll
    for (int t = 0; t < 4; t++) {
      v16bf vb = load_bfrag(Vbh + (t * 16) * S_LEN + kt, S_LEN);
      o[t] = wmma_bf16(pa, vb, o[t]);
    }
  }

#pragma unroll
  for (int t = 0; t < 4; t++) {
#pragma unroll
    for (int r = 0; r < 8; r++) {
      int s = q0 + half * 8 + r;
      float val = o[t][r] / lrun[r];
      AO[(size_t)(b * S_LEN + s) * DMODEL + h * DK + t * 16 + ln] = (bf16)val;
    }
  }
}

extern "C" void kernel_launch(void* const* d_in, const int* in_sizes, int n_in,
                              void* d_out, int out_size, void* d_ws, size_t ws_size,
                              hipStream_t stream) {
  (void)in_sizes; (void)n_in; (void)out_size; (void)ws_size;
  const float* x  = (const float*)d_in[0];
  const int*   mk = (const int*)d_in[1];
  const float* Wq = (const float*)d_in[2];
  const float* bq = (const float*)d_in[3];
  const float* Wk = (const float*)d_in[4];
  const float* bk = (const float*)d_in[5];
  const float* Wv = (const float*)d_in[6];
  const float* bv = (const float*)d_in[7];
  const float* Wo = (const float*)d_in[8];
  const float* bo = (const float*)d_in[9];

  char* ws = (char*)d_ws;
  size_t off = 0;
  auto carve = [&](size_t bytes) -> char* {
    char* p = ws + off;
    off += (bytes + 255) & ~(size_t)255;
    return p;
  };
  const size_t MTOT = 4ull * S_LEN;           // 8192 rows
  bf16* xb  = (bf16*)carve(MTOT * DMODEL * 2);
  bf16* wqt = (bf16*)carve((size_t)DMODEL * DMODEL * 2);
  bf16* wkt = (bf16*)carve((size_t)DMODEL * DMODEL * 2);
  bf16* wvt = (bf16*)carve((size_t)DMODEL * DMODEL * 2);
  bf16* wot = (bf16*)carve((size_t)DMODEL * DMODEL * 2);
  bf16* qb  = (bf16*)carve(MTOT * DMODEL * 2);
  bf16* kb  = (bf16*)carve(MTOT * DMODEL * 2);
  bf16* vt  = (bf16*)carve(MTOT * DMODEL * 2);
  bf16* ao  = (bf16*)carve(MTOT * DMODEL * 2);

  const int NX = (int)(MTOT * DMODEL);        // 8388608
  conv_f32_bf16<<<NX / 256, 256, 0, stream>>>(x, xb, NX);
  const int NW = DMODEL * DMODEL;             // 1048576
  transpose_conv<<<NW / 256, 256, 0, stream>>>(Wq, wqt);
  transpose_conv<<<NW / 256, 256, 0, stream>>>(Wk, wkt);
  transpose_conv<<<NW / 256, 256, 0, stream>>>(Wv, wvt);
  transpose_conv<<<NW / 256, 256, 0, stream>>>(Wo, wot);

  dim3 ggrid(64, 8);  // M/128, N/128
  gemm_kernel<0><<<ggrid, 256, 0, stream>>>(xb, wqt, bq, (void*)qb);
  gemm_kernel<0><<<ggrid, 256, 0, stream>>>(xb, wkt, bk, (void*)kb);
  gemm_kernel<1><<<ggrid, 256, 0, stream>>>(xb, wvt, bv, (void*)vt);

  attn_kernel<<<dim3(16, 64), 256, 0, stream>>>(qb, kb, vt, mk, ao);

  gemm_kernel<2><<<ggrid, 256, 0, stream>>>(ao, wot, bo, d_out);
}
